// GCN_63943473103101
// MI455X (gfx1250) — matrix operand order repaired and checked by reading, not compiled, not run on hardware
//
#include <hip/hip_runtime.h>
#include <hip/hip_bf16.h>

typedef _Float16 half_t;
typedef _Float16 v16h __attribute__((ext_vector_type(16)));
typedef float    v8f  __attribute__((ext_vector_type(8)));
typedef unsigned int u32x4 __attribute__((ext_vector_type(4)));
typedef int          i32x8 __attribute__((ext_vector_type(8)));
typedef int          i32x4 __attribute__((ext_vector_type(4)));

// ---------------------------------------------------------------------------
// f32 [K][N] row-major  ->  f16 [N][K] (transposed). One-off weight prep so
// GEMM B-operands are column-major: TDM tiles then land in LDS already in
// WMMA B-fragment order (each lane's 16 K-values contiguous).
// ---------------------------------------------------------------------------
__global__ void f32_to_f16_T_kernel(const float* __restrict__ src,
                                    half_t* __restrict__ dst, int K, int N) {
    int i = blockIdx.x * 256 + threadIdx.x;
    if (i < K * N) {
        int k = i / N, n = i % N;
        dst[(size_t)n * K + k] = (half_t)src[i];
    }
}

// ---------------------------------------------------------------------------
// Kernel 1: conv1 (1->16ch, k3 s1 p1) + ReLU + maxpool3(k3 s2 p1), fused.
// Avoids materializing the 536MB conv1 activation in HBM (67MB instead).
// ---------------------------------------------------------------------------
__global__ __launch_bounds__(256) void conv1_pool1_kernel(
    const float* __restrict__ x, const float* __restrict__ w,
    const float* __restrict__ bias, float* __restrict__ pool1) {
    __shared__ float s_in[5 * 34 * 34];
    __shared__ float s_conv[2 * 3 * 32 * 32];
    __shared__ float s_w[16 * 27];
    __shared__ float s_b[16];

    const int tid = threadIdx.x;
    const int bn  = blockIdx.x >> 4;
    const int pz  = blockIdx.x & 15;

    for (int i = tid; i < 432; i += 256) s_w[i] = w[i];
    if (tid < 16) s_b[tid] = bias[tid];

    const int izb = 2 * pz - 2;
    const float* xin = x + (size_t)bn * 32768;
    for (int i = tid; i < 5 * 34 * 34; i += 256) {
        int z   = i / (34 * 34);
        int rem = i % (34 * 34);
        int yy  = rem / 34 - 1;
        int xx  = rem % 34 - 1;
        int gz  = izb + z;
        float v = 0.f;
        if (gz >= 0 && gz < 32 && yy >= 0 && yy < 32 && xx >= 0 && xx < 32)
            v = xin[gz * 1024 + yy * 32 + xx];
        s_in[i] = v;
    }
    __syncthreads();

    float* out = pool1 + (size_t)bn * 65536;
    for (int c0 = 0; c0 < 16; c0 += 2) {
        for (int i = tid; i < 2 * 3 * 1024; i += 256) {
            int ocl = i / 3072;
            int rem = i % 3072;
            int p   = rem / 1024;
            int pos = rem & 1023;
            int cy  = pos >> 5, cx = pos & 31;
            int cz  = 2 * pz - 1 + p;
            float acc = -1e30f;
            if (cz >= 0 && cz < 32) {
                int oc = c0 + ocl;
                acc = s_b[oc];
                const float* wp = &s_w[oc * 27];
#pragma unroll
                for (int dz = 0; dz < 3; ++dz)
#pragma unroll
                    for (int dy = 0; dy < 3; ++dy)
#pragma unroll
                        for (int dx = 0; dx < 3; ++dx)
                            acc += wp[dz * 9 + dy * 3 + dx] *
                                   s_in[(p + dz) * 1156 + (cy + dy) * 34 + (cx + dx)];
                acc = fmaxf(acc, 0.f);
            }
            s_conv[i] = acc;
        }
        __syncthreads();
        for (int j = 0; j < 2; ++j) {
            int idx = tid + j * 256;
            int ocl = idx >> 8;
            int pos = idx & 255;
            int py = pos >> 4, px = pos & 15;
            float m = -1e30f;
#pragma unroll
            for (int p = 0; p < 3; ++p)
#pragma unroll
                for (int q = 0; q < 3; ++q)
#pragma unroll
                    for (int r = 0; r < 3; ++r) {
                        int cy = 2 * py - 1 + q;
                        int cx = 2 * px - 1 + r;
                        if (cy >= 0 && cx >= 0)
                            m = fmaxf(m, s_conv[ocl * 3072 + p * 1024 + cy * 32 + cx]);
                    }
            out[(size_t)(c0 + ocl) * 4096 + pz * 256 + py * 16 + px] = m;
        }
        __syncthreads();
    }
}

// ---------------------------------------------------------------------------
// Kernel 2: conv2 (16->32ch, k3 s2 p1) + ReLU. Block = (bn, oz).
// ---------------------------------------------------------------------------
__global__ __launch_bounds__(256) void conv2_kernel(
    const float* __restrict__ pool1, const float* __restrict__ w,
    const float* __restrict__ bias, float* __restrict__ out) {
    __shared__ float s_in[16 * 3 * 256];
    __shared__ float s_w[8 * 432];
    __shared__ float s_b[32];

    const int tid = threadIdx.x;
    const int bn  = blockIdx.x >> 3;
    const int oz  = blockIdx.x & 7;

    if (tid < 32) s_b[tid] = bias[tid];
    const float* pin = pool1 + (size_t)bn * 65536;
    for (int i = tid; i < 16 * 3 * 256; i += 256) {
        int ic  = i / 768;
        int rem = i % 768;
        int dz  = rem / 256;
        int pos = rem & 255;
        int iz  = 2 * oz - 1 + dz;
        s_in[i] = (iz >= 0 && iz < 16) ? pin[ic * 4096 + iz * 256 + pos] : 0.f;
    }
    __syncthreads();

    const int pos = tid & 63;
    const int sg  = tid >> 6;
    const int oy  = pos >> 3, ox = pos & 7;
    float* o = out + (size_t)bn * 16384 + oz * 64 + oy * 8 + ox;

    for (int chunk = 0; chunk < 4; ++chunk) {
        for (int i = tid; i < 8 * 432; i += 256) s_w[i] = w[chunk * (8 * 432) + i];
        __syncthreads();
        float acc0 = 0.f, acc1 = 0.f;
        const int wo0 = (sg * 2) * 432, wo1 = (sg * 2 + 1) * 432;
        for (int ic = 0; ic < 16; ++ic) {
#pragma unroll
            for (int dz = 0; dz < 3; ++dz)
#pragma unroll
                for (int dy = 0; dy < 3; ++dy) {
                    int iy = 2 * oy - 1 + dy;
                    if (iy < 0 || iy >= 16) continue;
#pragma unroll
                    for (int dx = 0; dx < 3; ++dx) {
                        int ix = 2 * ox - 1 + dx;
                        if (ix < 0 || ix >= 16) continue;
                        float v = s_in[ic * 768 + dz * 256 + iy * 16 + ix];
                        int wb  = ic * 27 + dz * 9 + dy * 3 + dx;
                        acc0 += v * s_w[wo0 + wb];
                        acc1 += v * s_w[wo1 + wb];
                    }
                }
        }
        int oc0 = chunk * 8 + sg * 2;
        o[(size_t)oc0 * 512]       = fmaxf(acc0 + s_b[oc0], 0.f);
        o[(size_t)(oc0 + 1) * 512] = fmaxf(acc1 + s_b[oc0 + 1], 0.f);
        __syncthreads();
    }
}

// ---------------------------------------------------------------------------
// Kernel 3: maxpool3(k3 s2 p1) -> node features, f16.
// ---------------------------------------------------------------------------
__global__ void pool2_feat_kernel(const float* __restrict__ conv2,
                                  half_t* __restrict__ featH) {
    int idx = blockIdx.x * 256 + threadIdx.x;
    int bn = idx >> 11;
    int f  = idx & 2047;
    int oc = f >> 6;
    int pz = (f >> 4) & 3, py = (f >> 2) & 3, px = f & 3;
    const float* c = conv2 + (size_t)bn * 16384 + (size_t)oc * 512;
    float m = -1e30f;
    for (int p = 0; p < 3; ++p) {
        int z = 2 * pz - 1 + p; if (z < 0 || z >= 8) continue;
        for (int q = 0; q < 3; ++q) {
            int y = 2 * py - 1 + q; if (y < 0 || y >= 8) continue;
            for (int r = 0; r < 3; ++r) {
                int xx = 2 * px - 1 + r; if (xx < 0 || xx >= 8) continue;
                m = fmaxf(m, c[z * 64 + y * 8 + xx]);
            }
        }
    }
    featH[idx] = (half_t)m;
}

// ---------------------------------------------------------------------------
// Kernel 4: GCN normalization  A_hat = D^-1/2 (A+I) D^-1/2  -> f16.
// ---------------------------------------------------------------------------
__global__ void adjnorm_kernel(const int* __restrict__ adj,
                               half_t* __restrict__ Ah) {
    __shared__ float dinv[64];
    int b = blockIdx.x;
    int i = threadIdx.x;
    const int* a = adj + b * 4096;
    int deg = 1;
    for (int j = 0; j < 64; ++j) deg += (a[i * 64 + j] != 0);
    dinv[i] = rsqrtf((float)deg);
    __syncthreads();
    half_t* o = Ah + b * 4096;
    float di = dinv[i];
    for (int j = 0; j < 64; ++j) {
        float v = (i == j) ? 1.f : (a[i * 64 + j] != 0 ? 1.f : 0.f);
        o[i * 64 + j] = (half_t)(v * di * dinv[j]);
    }
}

// ---------------------------------------------------------------------------
// TDM: DMA a [rows x cols] f16 tile (row stride = stride elements) from
// global memory into LDS (rows land contiguously). D# per CDNA5 ISA ch.8.
// Completion tracked by TENSORcnt (s_wait_tensorcnt).
// ---------------------------------------------------------------------------
__device__ __forceinline__ void tdm_load_tile_f16(const half_t* gsrc, void* lds,
                                                  int stride, int rows, int cols) {
    unsigned long long ga = (unsigned long long)gsrc;
    unsigned int la = (unsigned int)(unsigned long long)lds;  // flat addr[31:0] = LDS offset
    u32x4 g0;
    g0[0] = 1u;                                   // count=1, user descriptor
    g0[1] = la;                                   // lds_addr
    g0[2] = (unsigned int)ga;                     // global_addr[31:0]
    g0[3] = (unsigned int)((ga >> 32) & 0x01FFFFFFu) | (2u << 30);  // ga[56:32], type=2
    unsigned int td0 = (unsigned int)stride;      // tensor_dim0 = row length
    unsigned int td1 = 0x7FFFFFFFu;               // tensor_dim1: large (tiles in-bounds)
    i32x8 g1;
    g1[0] = (int)(1u << 16);                      // data_size = 1 -> 2 bytes
    g1[1] = (int)((td0 & 0xFFFFu) << 16);         // tensor_dim0[15:0]
    g1[2] = (int)((td0 >> 16) | ((td1 & 0xFFFFu) << 16));
    g1[3] = (int)((td1 >> 16) | ((unsigned int)cols << 16));  // tile_dim0 = cols
    g1[4] = (int)(unsigned int)rows;              // tile_dim1 = rows, tile_dim2 = 0
    g1[5] = (int)(unsigned int)stride;            // tensor_dim0_stride[31:0]
    g1[6] = 0;
    g1[7] = 0;
    i32x4 z4 = {0, 0, 0, 0};
    i32x8 z8 = {0, 0, 0, 0, 0, 0, 0, 0};
    __builtin_amdgcn_tensor_load_to_lds(g0, g1, z4, z4, z8, 0);
}

// ---------------------------------------------------------------------------
// Kernel 5: batched WMMA GEMM, C = A(f16,[M,K]) @ B(f16,[K,N]) (+bias,+relu).
// B is supplied TRANSPOSED: Bt[N][K] with row stride ldB elements. The TDM
// DMAs a 64(n) x 32(k) Bt tile into LDS (double buffered; next slice streams
// while this slice's WMMAs run). Because Bt rows are K-contiguous, each
// lane's B fragment is 32 contiguous LDS bytes -> two ds_load_b128, no
// gather/pack. All four B fragments are staged before the WMMA chain so a
// single dscnt wait covers them and the 4 WMMAs issue back-to-back.
// A fragments: K-pairs contiguous -> 2 clause-grouped global b128 loads.
// 4 waves/block, wave w owns rows m0=bx*64+w*16; block owns cols n0=by*64.
// TRANS_C stores C transposed (Ct[n][m]) to feed the next GEMM's B operand.
// ---------------------------------------------------------------------------
template <bool HAS_BIAS, bool RELU, bool STORE_H, bool STORE_F, bool TRANS_C>
__global__ __launch_bounds__(128) void wmma_gemm_kernel(
    const half_t* __restrict__ A, const half_t* __restrict__ Bt,
    half_t* __restrict__ Ch, float* __restrict__ Cf,
    const float* __restrict__ bias,
    int M, int N, int K, int ldB, int aBS, int bBS, int cBS) {
    __shared__ half_t sB[2][64 * 32];   // [n-row][k-col], rows contiguous

    const int wave = threadIdx.x >> 5;
    const int lane = threadIdx.x & 31;
    const int h    = lane >> 4;          // K-half selector
    const int l    = lane & 15;          // row (A frag) / col (B,D frags)
    const int m0   = blockIdx.x * 64 + wave * 16;
    const int n0   = blockIdx.y * 64;
    const int bz   = blockIdx.z;

    const half_t* Ab  = A  + (size_t)bz * aBS;
    const half_t* Btb = Bt + (size_t)bz * bBS + (size_t)n0 * ldB;

    v8f acc[4] = {{}, {}, {}, {}};
    const int mr = m0 + l;

    // prologue: stage first Bt tile (64 n-rows x 32 k-cols)
    if (wave == 0) {
        tdm_load_tile_f16(Btb, &sB[0][0], ldB, 64, 32);
        __builtin_amdgcn_s_wait_tensorcnt(0);
    }
    __syncthreads();

#pragma unroll 2
    for (int k0 = 0; k0 < K; k0 += 32) {
        const int buf = (k0 >> 5) & 1;
        const bool more = (k0 + 32) < K;
        if (wave == 0 && more)  // overlap next DMA with this slice's WMMAs
            tdm_load_tile_f16(Btb + (k0 + 32), &sB[buf ^ 1][0], ldB, 64, 32);

        // A fragment: 8 dwords (K-pairs contiguous; lowers to 2x b128)
        union { v16h v; unsigned int u[8]; } af;
        const unsigned int* Au = (const unsigned int*)(Ab + (size_t)mr * K + k0);
#pragma unroll
        for (int v = 0; v < 8; ++v)
            af.u[v] = Au[v + 8 * h];

        // stage all four B fragments (8x ds_load_b128), then WMMA chain
        union { v16h v; u32x4 q[2]; } bf[4];
#pragma unroll
        for (int t = 0; t < 4; ++t) {
            const u32x4* bp = (const u32x4*)&sB[buf][(t * 16 + l) * 32 + h * 16];
            bf[t].q[0] = bp[0];
            bf[t].q[1] = bp[1];
        }
#pragma unroll
        for (int t = 0; t < 4; ++t)
            acc[t] = __builtin_amdgcn_wmma_f32_16x16x32_f16(
                false, af.v, false, bf[t].v, (short)0, acc[t], false, false);

        if (wave == 0 && more) __builtin_amdgcn_s_wait_tensorcnt(0);
        __syncthreads();
    }

#pragma unroll
    for (int t = 0; t < 4; ++t) {
        const int n = n0 + t * 16 + l;
        float bv = 0.f;
        if (HAS_BIAS) bv = bias[n];
#pragma unroll
        for (int r = 0; r < 8; ++r) {
            int m = m0 + r + 8 * h;
            float vv = acc[t][r] + bv;
            if (RELU) vv = fmaxf(vv, 0.f);
            size_t off = TRANS_C ? ((size_t)n * M + m)
                                 : ((size_t)cBS * bz + (size_t)m * N + n);
            if (STORE_H) Ch[off] = (half_t)vv;
            if (STORE_F) Cf[off] = vv;
        }
    }
}

// ---------------------------------------------------------------------------
// Kernel 6: head. g[4,64,256] -> reshape[4,256,64].max(-1) -> fc1+relu ->
// fc2 -> softmax.
// ---------------------------------------------------------------------------
__global__ __launch_bounds__(256) void head_kernel(
    const float* __restrict__ G, const float* __restrict__ w1,
    const float* __restrict__ b1, const float* __restrict__ w2,
    const float* __restrict__ b2, float* __restrict__ out) {
    __shared__ float s_p[4 * 256];
    __shared__ float s_h[4 * 512];
    __shared__ float s_l[16];
    const int tid = threadIdx.x;

    for (int i = tid; i < 1024; i += 256) {
        int b = i >> 8;
        int k = i & 255;
        int n = k >> 2, j = k & 3;
        const float* g = G + (size_t)b * 16384 + n * 256 + j * 64;
        float m = g[0];
        for (int t = 1; t < 64; ++t) m = fmaxf(m, g[t]);
        s_p[i] = m;
    }
    __syncthreads();
    for (int i = tid; i < 2048; i += 256) {
        int b = i >> 9;
        int o = i & 511;
        float acc = b1[o];
        for (int k = 0; k < 256; ++k) acc += s_p[b * 256 + k] * w1[k * 512 + o];
        s_h[i] = fmaxf(acc, 0.f);
    }
    __syncthreads();
    if (tid < 16) {
        int b = tid >> 2, c = tid & 3;
        float acc = b2[c];
        for (int k = 0; k < 512; ++k) acc += s_h[b * 512 + k] * w2[k * 4 + c];
        s_l[tid] = acc;
    }
    __syncthreads();
    if (tid < 4) {
        float mx = s_l[tid * 4];
        for (int c = 1; c < 4; ++c) mx = fmaxf(mx, s_l[tid * 4 + c]);
        float e[4], sum = 0.f;
        for (int c = 0; c < 4; ++c) { e[c] = __expf(s_l[tid * 4 + c] - mx); sum += e[c]; }
        for (int c = 0; c < 4; ++c) out[tid * 4 + c] = e[c] / sum;
    }
}

// ---------------------------------------------------------------------------
extern "C" void kernel_launch(void* const* d_in, const int* in_sizes, int n_in,
                              void* d_out, int out_size, void* d_ws, size_t ws_size,
                              hipStream_t stream) {
    (void)in_sizes; (void)n_in; (void)out_size; (void)ws_size;
    const float* x       = (const float*)d_in[0];
    const int*   adj     = (const int*)d_in[1];
    const float* w_conv1 = (const float*)d_in[3];
    const float* b_conv1 = (const float*)d_in[4];
    const float* w_conv2 = (const float*)d_in[5];
    const float* b_conv2 = (const float*)d_in[6];
    const float* w_g1    = (const float*)d_in[7];
    const float* b_g1    = (const float*)d_in[8];
    const float* w_g2    = (const float*)d_in[9];
    const float* b_g2    = (const float*)d_in[10];
    const float* w_g3    = (const float*)d_in[11];
    const float* b_g3    = (const float*)d_in[12];
    const float* w_fc1   = (const float*)d_in[13];
    const float* b_fc1   = (const float*)d_in[14];
    const float* w_fc2   = (const float*)d_in[15];
    const float* b_fc2   = (const float*)d_in[16];
    float* out = (float*)d_out;

    char* ws = (char*)d_ws;
    size_t off = 0;
    auto carve = [&](size_t bytes) {
        void* p = ws + off;
        off += (bytes + 255) & ~(size_t)255;
        return p;
    };
    float*  pool1 = (float*)carve((size_t)256 * 16 * 4096 * 4);   // 67 MB
    float*  c2    = (float*)carve((size_t)256 * 32 * 512 * 4);    // 16 MB
    half_t* featH = (half_t*)carve((size_t)256 * 2048 * 2);
    half_t* Ah    = (half_t*)carve((size_t)4 * 64 * 64 * 2);
    half_t* Wg1t  = (half_t*)carve((size_t)1024 * 2048 * 2);      // [N][K]
    half_t* Wg2t  = (half_t*)carve((size_t)512 * 1024 * 2);
    half_t* Wg3t  = (half_t*)carve((size_t)256 * 512 * 2);
    half_t* Y1t   = (half_t*)carve((size_t)1024 * 256 * 2);       // [N][M]
    half_t* X1h   = (half_t*)carve((size_t)256 * 1024 * 2);       // [M][N]
    half_t* Y2t   = (half_t*)carve((size_t)512 * 256 * 2);
    half_t* X2h   = (half_t*)carve((size_t)256 * 512 * 2);
    half_t* Y3t   = (half_t*)carve((size_t)256 * 256 * 2);
    float*  G     = (float*)carve((size_t)256 * 256 * 4);

    // weight prep: f32 [K][N] -> f16 [N][K]
    f32_to_f16_T_kernel<<<(2048 * 1024 + 255) / 256, 256, 0, stream>>>(w_g1, Wg1t, 2048, 1024);
    f32_to_f16_T_kernel<<<(1024 * 512 + 255) / 256, 256, 0, stream>>>(w_g2, Wg2t, 1024, 512);
    f32_to_f16_T_kernel<<<(512 * 256 + 255) / 256, 256, 0, stream>>>(w_g3, Wg3t, 512, 256);

    conv1_pool1_kernel<<<256 * 16, 256, 0, stream>>>(x, w_conv1, b_conv1, pool1);
    conv2_kernel<<<256 * 8, 256, 0, stream>>>(pool1, w_conv2, b_conv2, c2);
    pool2_feat_kernel<<<(256 * 2048) / 256, 256, 0, stream>>>(c2, featH);
    adjnorm_kernel<<<4, 64, 0, stream>>>(adj, Ah);

    // Layer 1: Y1t = (feat @ Wg1)^T ; X1 = relu(Ah @ Y1 + b_g1)
    wmma_gemm_kernel<false, false, true, false, true><<<dim3(4, 1024 / 64, 1), 128, 0, stream>>>(
        featH, Wg1t, Y1t, nullptr, nullptr, 256, 1024, 2048, 2048, 0, 0, 0);
    wmma_gemm_kernel<true, true, true, false, false><<<dim3(1, 1024 / 64, 4), 128, 0, stream>>>(
        Ah, Y1t, X1h, nullptr, b_g1, 64, 1024, 64, 256, 4096, 64, 64 * 1024);

    // Layer 2
    wmma_gemm_kernel<false, false, true, false, true><<<dim3(4, 512 / 64, 1), 128, 0, stream>>>(
        X1h, Wg2t, Y2t, nullptr, nullptr, 256, 512, 1024, 1024, 0, 0, 0);
    wmma_gemm_kernel<true, true, true, false, false><<<dim3(1, 512 / 64, 4), 128, 0, stream>>>(
        Ah, Y2t, X2h, nullptr, b_g2, 64, 512, 64, 256, 4096, 64, 64 * 512);

    // Layer 3 (no relu), f32 out for the head
    wmma_gemm_kernel<false, false, true, false, true><<<dim3(4, 256 / 64, 1), 128, 0, stream>>>(
        X2h, Wg3t, Y3t, nullptr, nullptr, 256, 256, 512, 512, 0, 0, 0);
    wmma_gemm_kernel<true, false, false, true, false><<<dim3(1, 256 / 64, 4), 128, 0, stream>>>(
        Ah, Y3t, nullptr, G, b_g3, 64, 256, 64, 256, 4096, 64, 64 * 256);

    head_kernel<<<1, 256, 0, stream>>>(G, w_fc1, b_fc1, w_fc2, b_fc2, out);
}
